// Decoder_88897233092702
// MI455X (gfx1250) — matrix operand order repaired and checked by reading, not compiled, not run on hardware
//
#include <hip/hip_runtime.h>
#include <hip/hip_bf16.h>
#include <stddef.h>

// ---------------- CDNA5 WMMA types ----------------
typedef __attribute__((ext_vector_type(16))) __bf16 v16bf;
typedef __attribute__((ext_vector_type(8)))  float  v8f;

#define Hq 1024
#define Vq 50257
#define Eq 512
#define Bq 128
#define Sq 512
#define XHLD 3584   // E + 2H + H   (emb | ctx | h) packed bf16 activations

// ---------------- ws layout (bytes) ----------------
#define OFF_XH     0u                         // B*3584 bf16        = 917504
#define OFF_WAE    917504u                    // 1024*2048 bf16     = 4 MB
#define OFF_AH     (917504u + 4194304u)       // B*H f32            = 512 KB
#define OFF_SCORES (OFF_AH + 524288u)         // B*S f32            = 256 KB
#define OFF_ATTN   (OFF_SCORES + 262144u)     // B*S f32            = 256 KB
#define OFF_GX     (OFF_ATTN + 262144u)       // B*3H f32           = 1.5 MB
#define OFF_GH     (OFF_GX + 1572864u)        // B*3H f32           = 1.5 MB
#define OFF_HNBF   (OFF_GH + 1572864u)        // B*H bf16           = 256 KB

// =====================================================================
// prep: embedding lookup + h -> bf16, packed into xh = [emb | ctx | h]
// =====================================================================
__global__ __launch_bounds__(256) void prep_kernel(
    const int* __restrict__ idx, const float* __restrict__ h,
    const float* __restrict__ embt, __bf16* __restrict__ xh) {
  const int b = blockIdx.x, tid = threadIdx.x;
  const int token = idx[b];
  for (int e = tid; e < Eq; e += 256)
    xh[(size_t)b * XHLD + e] = (__bf16)embt[(size_t)token * Eq + e];
  for (int k = tid; k < Hq; k += 256)
    xh[(size_t)b * XHLD + (Eq + 2 * Hq) + k] = (__bf16)h[(size_t)b * Hq + k];
}

// =====================================================================
// f32 -> bf16 elementwise (for W_ae)
// =====================================================================
__global__ __launch_bounds__(256) void cvt_bf16(
    const float* __restrict__ src, __bf16* __restrict__ dst, int n) {
  int i = blockIdx.x * 256 + threadIdx.x;
  if (i < n) dst[i] = (__bf16)src[i];
}

// =====================================================================
// Generic WMMA GEMM: C[m][n] = sum_k A[m][k] * W[n][k] + bias[n]
//   M fixed = 128 (8 waves, one 16-row M-tile per wave)
//   A: bf16 row-major (lda), W: f32 row-major [N][K] staged->bf16 in LDS
//   K must be a multiple of 512. N bounds-checked (vocab tail).
//   grid.x = ceil(N/64); each WG covers 4 N-tiles of 16.
// =====================================================================
__global__ __launch_bounds__(256) void gemm_bf16(
    const __bf16* __restrict__ A, int lda,
    const float* __restrict__ W,
    const float* __restrict__ bias,
    float* __restrict__ C, int ldc, int N, int K) {
  __shared__ __align__(32) __bf16 Wt[16][512];   // 16 KB B-staging
  const int tid  = threadIdx.x;
  const int lane = tid & 31;
  const int wave = tid >> 5;
  const int g    = lane >> 4;     // half-wave group
  const int ln   = lane & 15;
  const int m0   = wave * 16;     // this wave's M-tile

  for (int nt = 0; nt < 4; ++nt) {
    const int n0 = (blockIdx.x * 4 + nt) * 16;
    v8f acc = {};
    for (int kc = 0; kc < K; kc += 512) {
      __syncthreads();
      // stage W[n0..n0+15][kc..kc+511] -> bf16 LDS (float4 coalesced)
      for (int i = tid; i < 16 * 128; i += 256) {
        const int rowl = i >> 7;
        const int c4   = (i & 127) << 2;
        const int n    = n0 + rowl;
        float4 w = (n < N) ? *(const float4*)(W + (size_t)n * K + kc + c4)
                           : make_float4(0.f, 0.f, 0.f, 0.f);
        __bf16* d = &Wt[rowl][c4];
        d[0] = (__bf16)w.x; d[1] = (__bf16)w.y;
        d[2] = (__bf16)w.z; d[3] = (__bf16)w.w;
      }
      __syncthreads();
#pragma unroll 4
      for (int kt = 0; kt < 16; ++kt) {
        const int k0 = kc + kt * 32;
        v16bf a, bf;
        // A 16x32 bf16 fragment: lane-half g holds K {g*16..g*16+15}
        const __bf16* ap = A + (size_t)(m0 + ln) * lda + k0 + g * 16;
        ((uint4*)&a)[0] = *(const uint4*)ap;
        ((uint4*)&a)[1] = *(const uint4*)(ap + 8);
        // B 32x16 fragment: lane = n, half-wave g holds K {g*16..g*16+15}
        bf = *(const v16bf*)&Wt[ln][kt * 32 + g * 16];
        acc = __builtin_amdgcn_wmma_f32_16x16x32_bf16(
            false, a, false, bf, (short)0, acc, false, false);
      }
    }
    const int n = n0 + ln;
    if (n < N) {
      const float bv = bias ? bias[n] : 0.f;
#pragma unroll
      for (int r = 0; r < 8; ++r) {
        const int m = m0 + g * 8 + r;             // C layout: m = 8*g + vgpr
        C[(size_t)m * ldc + n] = acc[r] + bv;
      }
    }
  }
}

// =====================================================================
// Attention scores, fused:
//   scores[b][s] = w_score . tanh(ah[b] + b_ae + enc[b,s] @ W_ae.T)
// One WG per (b, 32-row s-tile): 128 KB LDS enc tile -> TWO WGs resident
// per WGP (320 KB), so one WG's f32->bf16 staging prologue overlaps the
// other's WMMA phase, and 16 waves/WGP hide the W_ae global-load stream.
// Waves: mt = wave>>2 (2 M-tiles), ncol = wave&3 (4 N-cols); 16
// N-iterations x 64 K-tiles of v_wmma_f32_16x16x32_bf16; epilogue does
// tanh + 16-lane shuffle reduce + ds_add_f32 into score accumulators.
// =====================================================================
__global__ __launch_bounds__(256) void attn_scores(
    const float* __restrict__ enc, const __bf16* __restrict__ Wae,
    const float* __restrict__ ah, const float* __restrict__ b_ae,
    const float* __restrict__ wsc, float* __restrict__ scores) {
  extern __shared__ char smem[];
  float*  sacc = (float*)smem;                 // 32 row accumulators
  __bf16* et   = (__bf16*)(smem + 256);        // [32][2048] bf16 enc tile

  const int b  = blockIdx.x >> 4;
  const int s0 = (blockIdx.x & 15) * 32;
  const int tid = threadIdx.x;
  if (tid < 32) sacc[tid] = 0.f;

  // stage 32 x 2048 f32 -> bf16 (float4 coalesced)
  const float* eb = enc + ((size_t)b * Sq + s0) * (2 * Hq);
  for (int i = tid; i < 32 * 512; i += 256) {
    const int row = i >> 9;
    const int c4  = (i & 511) << 2;
    float4 v = *(const float4*)(eb + (size_t)row * 2048 + c4);
    __bf16* d = et + (size_t)row * 2048 + c4;
    d[0] = (__bf16)v.x; d[1] = (__bf16)v.y;
    d[2] = (__bf16)v.z; d[3] = (__bf16)v.w;
  }
  __syncthreads();

  const int lane = tid & 31, wave = tid >> 5;
  const int g = lane >> 4, ln = lane & 15;
  const int mt   = wave >> 2;     // 0..1 : which 16-row s block
  const int ncol = wave & 3;      // 0..3 : which 16-col n block

  for (int niter = 0; niter < 16; ++niter) {
    const int n0 = niter * 64 + ncol * 16;
    v8f acc = {};
    const __bf16* arow = et  + (size_t)(mt * 16 + ln) * 2048 + g * 16;
    const __bf16* brow = Wae + (size_t)(n0 + ln) * 2048 + g * 16;
#pragma unroll 4
    for (int kt = 0; kt < 64; ++kt) {
      const int k0 = kt * 32;
      v16bf a, bf;
      ((uint4*)&a)[0] = *(const uint4*)(arow + k0);
      ((uint4*)&a)[1] = *(const uint4*)(arow + k0 + 8);
      bf = *(const v16bf*)(brow + k0);
      acc = __builtin_amdgcn_wmma_f32_16x16x32_bf16(
          false, a, false, bf, (short)0, acc, false, false);
    }
    // epilogue: tanh + dot with w_score, reduce over the 16 n-lanes
    const int n = n0 + ln;
    const float ahv = ah[(size_t)b * Hq + n] + b_ae[n];
    const float wv  = wsc[n];
#pragma unroll
    for (int r = 0; r < 8; ++r) {
      float v = tanhf(ahv + acc[r]) * wv;
      v += __shfl_xor(v, 1);
      v += __shfl_xor(v, 2);
      v += __shfl_xor(v, 4);
      v += __shfl_xor(v, 8);
      if (ln == 0) atomicAdd(&sacc[mt * 16 + g * 8 + r], v);  // ds_add_f32
    }
  }
  __syncthreads();
  if (tid < 32) scores[(size_t)b * Sq + s0 + tid] = sacc[tid];
}

// =====================================================================
// softmax over S=512, one block per batch row
// =====================================================================
__global__ __launch_bounds__(256) void softmax512(
    const float* __restrict__ scores, float* __restrict__ attn) {
  __shared__ float red[256];
  const int b = blockIdx.x, tid = threadIdx.x;
  const float a0 = scores[(size_t)b * Sq + tid];
  const float a1 = scores[(size_t)b * Sq + 256 + tid];
  red[tid] = fmaxf(a0, a1);
  __syncthreads();
  for (int s = 128; s > 0; s >>= 1) {
    if (tid < s) red[tid] = fmaxf(red[tid], red[tid + s]);
    __syncthreads();
  }
  const float m = red[0];
  __syncthreads();
  const float e0 = __expf(a0 - m), e1 = __expf(a1 - m);
  red[tid] = e0 + e1;
  __syncthreads();
  for (int s = 128; s > 0; s >>= 1) {
    if (tid < s) red[tid] += red[tid + s];
    __syncthreads();
  }
  const float inv = 1.f / red[0];
  attn[(size_t)b * Sq + tid]       = e0 * inv;
  attn[(size_t)b * Sq + 256 + tid] = e1 * inv;
}

// =====================================================================
// ctx[b][d] = sum_s attn[b][s] * enc[b][s][d]  -> bf16 into xh[:,512+d]
// grid = B*8, 256 threads: contiguous d for coalesced streaming reads
// =====================================================================
__global__ __launch_bounds__(256) void ctx_kernel(
    const float* __restrict__ enc, const float* __restrict__ attn,
    __bf16* __restrict__ xh) {
  const int b = blockIdx.x >> 3;
  const int d = ((blockIdx.x & 7) << 8) + threadIdx.x;
  const float* eb = enc + (size_t)b * Sq * (2 * Hq) + d;
  const float* ab = attn + (size_t)b * Sq;
  float acc = 0.f;
#pragma unroll 4
  for (int s = 0; s < Sq; ++s) acc += ab[s] * eb[(size_t)s * (2 * Hq)];
  xh[(size_t)b * XHLD + Eq + d] = (__bf16)acc;
}

// =====================================================================
// GRU gates (PyTorch r,z,n layout) -> h_new (f32 to d_out, bf16 to ws)
// =====================================================================
__global__ __launch_bounds__(256) void gru_gate(
    const float* __restrict__ gx, const float* __restrict__ gh,
    const float* __restrict__ h, float* __restrict__ hnew_out,
    __bf16* __restrict__ hnew_bf) {
  const int b = blockIdx.x;
  const float* gxb = gx + (size_t)b * 3 * Hq;
  const float* ghb = gh + (size_t)b * 3 * Hq;
  for (int j = threadIdx.x; j < Hq; j += 256) {
    const float r  = 1.f / (1.f + __expf(-(gxb[j] + ghb[j])));
    const float z  = 1.f / (1.f + __expf(-(gxb[Hq + j] + ghb[Hq + j])));
    const float nn = tanhf(gxb[2 * Hq + j] + r * ghb[2 * Hq + j]);
    const float hn = (1.f - z) * nn + z * h[(size_t)b * Hq + j];
    hnew_out[(size_t)b * Hq + j] = hn;
    hnew_bf [(size_t)b * Hq + j] = (__bf16)hn;
  }
}

// =====================================================================
extern "C" void kernel_launch(void* const* d_in, const int* in_sizes, int n_in,
                              void* d_out, int out_size, void* d_ws,
                              size_t ws_size, hipStream_t stream) {
  (void)in_sizes; (void)n_in; (void)out_size; (void)ws_size;
  const int*   dec_in  = (const int*)  d_in[0];
  const float* h       = (const float*)d_in[1];
  const float* enc     = (const float*)d_in[2];
  const float* embt    = (const float*)d_in[3];
  const float* W_ah    = (const float*)d_in[4];
  const float* b_ah    = (const float*)d_in[5];
  const float* W_ae    = (const float*)d_in[6];
  const float* b_ae    = (const float*)d_in[7];
  const float* w_score = (const float*)d_in[8];
  const float* W_ih    = (const float*)d_in[9];
  const float* W_hh    = (const float*)d_in[10];
  const float* b_ih    = (const float*)d_in[11];
  const float* b_hh    = (const float*)d_in[12];
  const float* W_out   = (const float*)d_in[13];
  const float* b_out   = (const float*)d_in[14];

  float* out_logits = (float*)d_out;                       // [B, 1, V]
  float* out_h      = out_logits + (size_t)Bq * Vq;        // [1, B, H]

  char* ws = (char*)d_ws;
  __bf16* xh      = (__bf16*)(ws + OFF_XH);
  __bf16* wae_bf  = (__bf16*)(ws + OFF_WAE);
  float*  ah      = (float*) (ws + OFF_AH);
  float*  scores  = (float*) (ws + OFF_SCORES);
  float*  attn    = (float*) (ws + OFF_ATTN);
  float*  gx      = (float*) (ws + OFF_GX);
  float*  gh      = (float*) (ws + OFF_GH);
  __bf16* hnew_bf = (__bf16*)(ws + OFF_HNBF);

  // 1) embed + h->bf16 into packed activations
  prep_kernel<<<Bq, 256, 0, stream>>>(dec_in, h, embt, xh);
  // 2) W_ae -> bf16 (L2-resident WMMA B operand)
  cvt_bf16<<<(Hq * 2 * Hq + 255) / 256, 256, 0, stream>>>(
      W_ae, wae_bf, Hq * 2 * Hq);
  // 3) ah = h @ W_ah.T + b_ah   (M=128,N=1024,K=1024)
  gemm_bf16<<<Hq / 64, 256, 0, stream>>>(
      xh + (Eq + 2 * Hq), XHLD, W_ah, b_ah, ah, Hq, Hq, Hq);
  // 4) fused attention scores (274 GFLOP bf16 WMMA, 128KB LDS enc tiles,
  //    2 WGs/WGP so staging overlaps WMMA)
  attn_scores<<<Bq * (Sq / 32), 256, 256 + 32 * 2048 * 2, stream>>>(
      enc, wae_bf, ah, b_ae, w_score, scores);
  // 5) softmax over S
  softmax512<<<Bq, 256, 0, stream>>>(scores, attn);
  // 6) ctx = attn @ enc  -> bf16 into xh[:, E:E+2H]
  ctx_kernel<<<Bq * 8, 256, 0, stream>>>(enc, attn, xh);
  // 7) gx = [emb|ctx] @ W_ih.T + b_ih   (K=2560)
  gemm_bf16<<<3 * Hq / 64, 256, 0, stream>>>(
      xh, XHLD, W_ih, b_ih, gx, 3 * Hq, 3 * Hq, Eq + 2 * Hq);
  // 8) gh = h @ W_hh.T + b_hh           (K=1024)
  gemm_bf16<<<3 * Hq / 64, 256, 0, stream>>>(
      xh + (Eq + 2 * Hq), XHLD, W_hh, b_hh, gh, 3 * Hq, 3 * Hq, Hq);
  // 9) GRU gates -> h_new
  gru_gate<<<Bq, 256, 0, stream>>>(gx, gh, h, out_h, hnew_bf);
  // 10) logits = h_new @ W_out.T + b_out  (N=50257, tail-guarded)
  gemm_bf16<<<(Vq + 63) / 64, 256, 0, stream>>>(
      hnew_bf, Hq, W_out, b_out, out_logits, Vq, Vq, Hq);
}
